// BertCRFTkModel_48696339202403
// MI455X (gfx1250) — compile-verified, matrix operand order checked by reading, not run on hardware
//
#include <hip/hip_runtime.h>
#include <hip/hip_bf16.h>
#include <math.h>

// CRF forward (logsumexp scan via WMMA on exp-transformed scores) + Viterbi.
// B=128, T=2048, L=32.  Emission tiles double-buffered into LDS with
// gfx1250 async-to-LDS ops (ASYNCcnt) to overlap HBM/L2 latency with the
// per-step max/exp/WMMA work of the serial scan.

#define BB 128
#define TT 2048
#define LL 32
#define TS (TT - 1)   // number of emission steps (logits[:,1:] for llh)

typedef __attribute__((ext_vector_type(16))) _Float16 v16h;
typedef __attribute__((ext_vector_type(8)))  float    v8f;

// ---- gfx1250 async global->LDS helpers (ASYNCcnt path) --------------------
__device__ __forceinline__ void async_lds_b128(unsigned lds_addr, unsigned voff,
                                               const void* sbase) {
    asm volatile("global_load_async_to_lds_b128 %0, %1, %2"
                 :: "v"(lds_addr), "v"(voff), "s"(sbase) : "memory");
}
__device__ __forceinline__ void async_lds_b32(unsigned lds_addr, unsigned voff,
                                              const void* sbase) {
    asm volatile("global_load_async_to_lds_b32 %0, %1, %2"
                 :: "v"(lds_addr), "v"(voff), "s"(sbase) : "memory");
}
__device__ __forceinline__ void wait_async0() {
    asm volatile("s_wait_asynccnt 0x0" ::: "memory");
}
__device__ __forceinline__ unsigned lds_off32(const void* p) {
    // Flat shared-aperture addresses keep the LDS byte offset in the low 32b.
    return (unsigned)(unsigned long long)p;
}

// ---------------------------------------------------------------------------
// Forward-algorithm scan. One wave (32 lanes) per 16-batch tile; 8 blocks.
// Per step: score(D-layout regs) -> LDS -> row max -> A fragment
// P = exp(score - mx) in f16 -> two v_wmma_f32_16x16x32_f16 against the
// constant B fragments E = exp(trans) -> next = em + mx + log(D), masked.
// Emission tiles (16x32 f32 = 2KB/step) arrive via async-to-LDS double buffer.
// ---------------------------------------------------------------------------
__global__ void __launch_bounds__(32)
crf_forward_scan(const float* __restrict__ logits,
                 const int*   __restrict__ crf_mask,
                 const float* __restrict__ start_t,
                 const float* __restrict__ end_t,
                 const float* __restrict__ trans,
                 float* __restrict__ den_out)   // per-batch denominator
{
    // LDS layout: sc[16][32] f32 (2048B) | mxs[16] f32 (64B) | emb[2][16][32]
    __shared__ __align__(16) unsigned char lds_raw[2048 + 64 + 2 * 2048];
    float* sc  = (float*)lds_raw;                 // score tile
    float* mxs = (float*)(lds_raw + 2048);        // per-row max
    float* emb = (float*)(lds_raw + 2112);        // emission double buffer
    const unsigned lbase   = lds_off32(lds_raw);
    const unsigned emb_off = lbase + 2112;

    const int lane = threadIdx.x;
    const int n    = lane & 15;    // column within 16-wide N tile
    const int hi   = lane >> 4;    // half-wave select
    const int bb   = blockIdx.x * 16;

    // Constant B fragments: E[k][n] = exp(trans[k][n]), 32x16 f16 per tile.
    // 16-bit matrix K map: lanes 0-15 -> K {0..7, 16..23}; lanes 16-31 -> +8.
    v16h bf0, bf1;
#pragma unroll
    for (int e = 0; e < 16; ++e) {
        int k = ((e < 8) ? e : e + 8) + hi * 8;
        bf0[e] = (_Float16)__expf(trans[k * LL + n]);
        bf1[e] = (_Float16)__expf(trans[k * LL + n + 16]);
    }

    // Init: score0 = start + emissions[:,0] = start + logits[:,1,:]
    // D-layout: register r holds row m = r + 8*hi, column n (tile0)/n+16 (tile1).
    float sc0[8], sc1[8];
#pragma unroll
    for (int r = 0; r < 8; ++r) {
        int b = bb + r + hi * 8;
        sc0[r] = start_t[n]      + logits[((size_t)b * TT + 1) * LL + n];
        sc1[r] = start_t[n + 16] + logits[((size_t)b * TT + 1) * LL + n + 16];
    }

    // Prologue: async-fill emission buffer 1 with logits[:, 2, :] (for s=1).
    // Tile = 512 floats = 128 x 16B chunks; 4 chunks per lane.
#pragma unroll
    for (int i = 0; i < 4; ++i) {
        int c   = lane + 32 * i;        // chunk id: row = c>>3, 16B col = c&7
        int row = c >> 3;
        unsigned voff = (((unsigned)(bb + row) * TT + 2) * LL + (c & 7) * 4) * 4u;
        async_lds_b128(emb_off + 1 * 2048 + c * 16, voff, logits);
    }

    for (int s = 1; s < TS; ++s) {
        const int cur = s & 1;
        const int nxt = (s + 1) & 1;

        // Mirror score to LDS for cross-lane access.
#pragma unroll
        for (int r = 0; r < 8; ++r) {
            int m = r + hi * 8;
            sc[m * LL + n]      = sc0[r];
            sc[m * LL + n + 16] = sc1[r];
        }
        __syncthreads();

        // Row maxima (lanes 0..15, one row each).
        if (lane < 16) {
            float m0 = sc[lane * LL];
            for (int k = 1; k < LL; ++k) m0 = fmaxf(m0, sc[lane * LL + k]);
            mxs[lane] = m0;
        }
        __syncthreads();

        // A fragment: row m = n (A-matrix: lane holds row lane&15).
        float myMx = mxs[n];
        v16h a;
#pragma unroll
        for (int e = 0; e < 16; ++e) {
            int k = ((e < 8) ? e : e + 8) + hi * 8;
            a[e] = (_Float16)__expf(sc[n * LL + k] - myMx);
        }

        v8f c = {};
        v8f d0 = __builtin_amdgcn_wmma_f32_16x16x32_f16(
            false, a, false, bf0, (short)0, c, false, false);
        v8f d1 = __builtin_amdgcn_wmma_f32_16x16x32_f16(
            false, a, false, bf1, (short)0, c, false, false);

        // Wait for this step's emission tile, then consume it from LDS.
        wait_async0();

        // next = em + mx + log(sum); masked select per batch row.
#pragma unroll
        for (int r = 0; r < 8; ++r) {
            int m = r + hi * 8;
            int b = bb + m;
            float em0 = emb[cur * 512 + m * LL + n];
            float em1 = emb[cur * 512 + m * LL + n + 16];
            int   mk  = crf_mask[(size_t)b * TT + s + 1];
            float mr  = mxs[m];
            float n0  = em0 + mr + __logf(d0[r]);
            float n1  = em1 + mr + __logf(d1[r]);
            sc0[r] = mk ? n0 : sc0[r];
            sc1[r] = mk ? n1 : sc1[r];
        }

        // Prefetch next step's emission tile (logits[:, s+2, :]) async to LDS.
        if (s + 1 < TS) {
#pragma unroll
            for (int i = 0; i < 4; ++i) {
                int c   = lane + 32 * i;
                int row = c >> 3;
                unsigned voff =
                    (((unsigned)(bb + row) * TT + s + 2) * LL + (c & 7) * 4) * 4u;
                async_lds_b128(emb_off + nxt * 2048 + c * 16, voff, logits);
            }
        }
        __syncthreads();
    }

    // Denominator: logsumexp(score + end) per batch row.
#pragma unroll
    for (int r = 0; r < 8; ++r) {
        int m = r + hi * 8;
        sc[m * LL + n]      = sc0[r] + end_t[n];
        sc[m * LL + n + 16] = sc1[r] + end_t[n + 16];
    }
    __syncthreads();
    if (lane < 16) {
        float m0 = sc[lane * LL];
        for (int k = 1; k < LL; ++k) m0 = fmaxf(m0, sc[lane * LL + k]);
        float ss = 0.0f;
        for (int k = 0; k < LL; ++k) ss += __expf(sc[lane * LL + k] - m0);
        den_out[bb + lane] = m0 + __logf(ss);
    }
}

// ---------------------------------------------------------------------------
// Numerator: gold-path score per batch (one thread per batch).
// ---------------------------------------------------------------------------
__global__ void crf_numerator(const float* __restrict__ logits,
                              const int*   __restrict__ labels,
                              const int*   __restrict__ crf_mask,
                              const float* __restrict__ start_t,
                              const float* __restrict__ end_t,
                              const float* __restrict__ trans,
                              float* __restrict__ num_out)
{
    int b = blockIdx.x * blockDim.x + threadIdx.x;
    if (b >= BB) return;
    int prev = labels[(size_t)b * TT + 1] * crf_mask[(size_t)b * TT + 1];
    float num = start_t[prev] + logits[((size_t)b * TT + 1) * LL + prev];
    int last = prev;
    for (int s = 1; s < TS; ++s) {
        if (s + 8 < TS)
            __builtin_prefetch(&logits[((size_t)b * TT + s + 9) * LL], 0, 0);
        int mk = crf_mask[(size_t)b * TT + s + 1];
        int tg = labels[(size_t)b * TT + s + 1] * mk;
        float e = logits[((size_t)b * TT + s + 1) * LL + tg];
        if (mk) { num += trans[prev * LL + tg] + e; last = tg; }
        prev = tg;
    }
    num += end_t[last];
    num_out[b] = num;
}

// ---------------------------------------------------------------------------
// Viterbi forward: one wave per batch, lane = next label j. LDS-broadcast
// score vector each step; emission row async-double-buffered into LDS;
// uint8 argmax history in workspace.
// ---------------------------------------------------------------------------
__global__ void __launch_bounds__(32)
viterbi_scan(const float* __restrict__ logits,
             const float* __restrict__ start_t,
             const float* __restrict__ end_t,
             const float* __restrict__ trans,
             unsigned char* __restrict__ hist,
             int* __restrict__ last_out)
{
    // LDS: s[32] f32 (128B) | em[2][32] f32 (256B)
    __shared__ __align__(16) unsigned char lds_raw[128 + 256];
    float* s  = (float*)lds_raw;
    float* em = (float*)(lds_raw + 128);
    const unsigned em_off = lds_off32(lds_raw) + 128;

    const int j = threadIdx.x;
    const int b = blockIdx.x;

    float tc[LL];                           // column j of transitions
#pragma unroll
    for (int i = 0; i < LL; ++i) tc[i] = trans[i * LL + j];

    float score = start_t[j] + logits[(size_t)b * TT * LL + j];

    // Prologue: async-fill buffer 1 with emission row t=1.
    async_lds_b32(em_off + 1 * 128 + j * 4,
                  (((unsigned)b * TT + 1) * LL + j) * 4u, logits);

    for (int t = 1; t < TT; ++t) {
        const int cur = t & 1;
        s[j] = score;
        __syncthreads();
        wait_async0();
        float emv = em[cur * LL + j];
        // Prefetch emission row t+1 into the other buffer.
        if (t + 1 < TT) {
            async_lds_b32(em_off + ((t + 1) & 1) * 128 + j * 4,
                          (((unsigned)b * TT + t + 1) * LL + j) * 4u, logits);
        }
        float best = s[0] + tc[0];
        int arg = 0;
#pragma unroll
        for (int i = 1; i < LL; ++i) {
            float v = s[i] + tc[i];
            if (v > best) { best = v; arg = i; }
        }
        score = best + emv;
        hist[((size_t)b * (TT - 1) + (t - 1)) * LL + j] = (unsigned char)arg;
        __syncthreads();
    }

    s[j] = score + end_t[j];
    __syncthreads();
    if (j == 0) {
        float best = s[0]; int arg = 0;
        for (int i = 1; i < LL; ++i)
            if (s[i] > best) { best = s[i]; arg = i; }
        last_out[b] = arg;
    }
}

// ---------------------------------------------------------------------------
// Viterbi backtrack: one thread per batch; decoded tags written as floats.
// ---------------------------------------------------------------------------
__global__ void viterbi_backtrack(const unsigned char* __restrict__ hist,
                                  const int* __restrict__ last,
                                  float* __restrict__ out_tags)
{
    int b = blockIdx.x * blockDim.x + threadIdx.x;
    if (b >= BB) return;
    int tag = last[b];
    out_tags[(size_t)b * TT + (TT - 1)] = (float)tag;
    for (int t = TT - 2; t >= 0; --t) {
        tag = hist[((size_t)b * (TT - 1) + t) * LL + tag];
        out_tags[(size_t)b * TT + t] = (float)tag;
    }
}

// ---------------------------------------------------------------------------
// Deterministic final reduction: loss = sum_b (den_b - num_b).
// ---------------------------------------------------------------------------
__global__ void reduce_loss(const float* __restrict__ den,
                            const float* __restrict__ num,
                            float* __restrict__ out)
{
    if (threadIdx.x == 0 && blockIdx.x == 0) {
        float acc = 0.0f;
        for (int b = 0; b < BB; ++b) acc += den[b] - num[b];
        out[0] = acc;
    }
}

extern "C" void kernel_launch(void* const* d_in, const int* in_sizes, int n_in,
                              void* d_out, int out_size, void* d_ws, size_t ws_size,
                              hipStream_t stream) {
    const float* logits   = (const float*)d_in[0];
    const int*   labels   = (const int*)  d_in[1];
    const int*   crf_mask = (const int*)  d_in[2];
    const float* start_t  = (const float*)d_in[3];
    const float* end_t    = (const float*)d_in[4];
    const float* trans    = (const float*)d_in[5];
    float* out = (float*)d_out;

    // Workspace layout
    const size_t HIST_BYTES = (size_t)BB * (TT - 1) * LL;       // 8,384,512 B
    unsigned char* hist = (unsigned char*)d_ws;
    int*   last = (int*)  ((char*)d_ws + HIST_BYTES);           // 128 ints
    float* den  = (float*)((char*)d_ws + HIST_BYTES + 512);     // 128 floats
    float* num  = (float*)((char*)d_ws + HIST_BYTES + 1024);    // 128 floats

    crf_forward_scan<<<dim3(BB / 16), dim3(32), 0, stream>>>(
        logits, crf_mask, start_t, end_t, trans, den);
    crf_numerator<<<dim3(4), dim3(32), 0, stream>>>(
        logits, labels, crf_mask, start_t, end_t, trans, num);
    viterbi_scan<<<dim3(BB), dim3(32), 0, stream>>>(
        logits, start_t, end_t, trans, hist, last);
    viterbi_backtrack<<<dim3(4), dim3(32), 0, stream>>>(hist, last, out + 1);
    reduce_loss<<<dim3(1), dim3(1), 0, stream>>>(den, num, out);
}
